// MambaEHRModel_57793079935343
// MI455X (gfx1250) — matrix operand order repaired and checked by reading, not compiled, not run on hardware
//
#include <hip/hip_runtime.h>
#include <hip/hip_bf16.h>

// ---------------- model constants ----------------
#define B_    8
#define L_    512
#define DM    132      // d_model
#define DI    264      // d_inner
#define DS    16       // d_state
#define DTR   9        // dt_rank
#define NLAY  4
#define NMAM  5
#define NH    6
#define HD    22       // head dim = 132/6

#define KP_DM  160     // 132 padded to x32 (K dims)
#define KP_DI  288     // 264 padded to x32

// N dims padded to x64 so every wave owns exactly four 16x16 tiles (branch-free)
#define NP_IN  576     // 528 -> 576
#define NP_XP  64      // 41  -> 64
#define NP_OP  192     // 132 -> 192
#define NP_QKV 448     // 396 -> 448

// real (stored) widths
#define NR_IN  528
#define NR_XP  48      // dbc buffer keeps 48 cols (41 used)
#define NR_OP  132
#define NR_QKV 400     // qkv buffer keeps 400 cols (396 used)

#define RM   4096      // rows per mamba = B_*L_
#define ROWS 20480     // NMAM*RM

typedef __attribute__((ext_vector_type(16))) _Float16 v16h;
typedef __attribute__((ext_vector_type(8)))  _Float16 v8h;
typedef __attribute__((ext_vector_type(8)))  float    v8f;

// ---------------- weight packing: f32 -> f16, zero-padded ----------------
__global__ void pack_w_f16(const float* __restrict__ src, _Float16* __restrict__ dst,
                           int G, int Rg, int Rgp, int C, int Cp) {
    long total = (long)G * Rgp * Cp;
    for (long i = (long)blockIdx.x * blockDim.x + threadIdx.x; i < total;
         i += (long)gridDim.x * blockDim.x) {
        int  c = (int)(i % Cp);
        long t = i / Cp;
        int  r = (int)(t % Rgp);
        int  g = (int)(t / Rgp);
        float v = (r < Rg && c < C) ? src[((long)g * Rg + r) * C + c] : 0.f;
        dst[i] = (_Float16)v;
    }
}

__global__ void pack_bias_f32(const float* __restrict__ src, float* __restrict__ dst,
                              int n, int npad) {
    int i = blockIdx.x * blockDim.x + threadIdx.x;
    if (i < npad) dst[i] = (i < n) ? src[i] : 0.f;
}

// ---------------- act[m][b][l][d] = x[b][d][l] * mask[b][d][l] ----------------
__global__ void init_act(const float* __restrict__ x, const float* __restrict__ mask,
                         float* __restrict__ act) {
    long total = (long)NMAM * RM * DM;
    for (long i = (long)blockIdx.x * blockDim.x + threadIdx.x; i < total;
         i += (long)gridDim.x * blockDim.x) {
        int  d = (int)(i % DM);
        long t = i / DM;
        int  l = (int)(t % L_);
        t /= L_;
        int  b = (int)(t % B_);
        long s = ((long)b * DM + d) * L_ + l;
        act[i] = x[s] * mask[s];
    }
}

// ---------------- rmsnorm (or plain) + f16 pack to KP_DM cols ----------------
// grid.x = ROWS rows, block = 160 threads
__global__ void rmsnorm_pack(const float* __restrict__ act,
                             const float* __restrict__ normw, long normw_mstride,
                             _Float16* __restrict__ out, int do_norm) {
    __shared__ float red[KP_DM];
    int row = blockIdx.x;
    int m   = row >> 12;             // row / 4096
    int d   = threadIdx.x;
    const float* arow = act + (long)row * DM;
    float v = (d < DM) ? arow[d] : 0.f;
    float scale = 1.f;
    if (do_norm) {
        red[d] = v * v;
        __syncthreads();
        if (d == 0) {
            float s = 0.f;
            for (int i = 0; i < DM; ++i) s += red[i];
            red[0] = s;
        }
        __syncthreads();
        scale = rsqrtf(red[0] * (1.f / DM) + 1e-5f);
        float w = (d < DM) ? normw[(long)m * normw_mstride + d] : 0.f;
        v = v * scale * w;
    }
    out[(long)row * KP_DM + d] = (_Float16)((d < DM) ? v : 0.f);
}

// ---------------- register-blocked WMMA GEMM: C = A(f16,MxK) * W(f16,NxK)^T -------
// One wave per block (wave32). Each wave owns a 16x64 strip = exactly four 16x16
// tiles (N is padded to x64 with zero weight rows), so the K-loop is completely
// branch-free and EXEC stays all-ones through every v_wmma (ISA 7.12). The A
// fragment is loaded once per K-step and reused by 4 WMMAs (4x arithmetic
// intensity vs 1-tile waves). Store-side col<Nreal guard is after all WMMAs.
// grid = (M/16, Npad/64, NMAM), block = 32.
__global__ void gemm_wmma(const _Float16* __restrict__ A, long sAm,
                          const _Float16* __restrict__ W, long sWm,
                          float* __restrict__ C, long sCm,
                          int ldc, int Nreal, int K,
                          const float* __restrict__ bias, int add_res) {
    int lane = threadIdx.x & 31;
    int n0 = blockIdx.y * 64;
    int m0 = blockIdx.x * 16;
    int l  = lane & 15;
    int hi = lane >> 4;

    const _Float16* Ar = A + blockIdx.z * sAm + (long)(m0 + l) * K;
    const _Float16* Wr = W + blockIdx.z * sWm + (long)(n0 + l) * K;
    long wjstride = (long)16 * K;             // next N-tile of the weight matrix

    v8f acc0 = {0.f,0.f,0.f,0.f,0.f,0.f,0.f,0.f};
    v8f acc1 = acc0, acc2 = acc0, acc3 = acc0;

    for (int k0 = 0; k0 < K; k0 += 32) {
        const _Float16* ap = Ar + k0 + hi * 16;
        v8h alo = *(const v8h*)(ap);
        v8h ahi = *(const v8h*)(ap + 8);
        v16h av = __builtin_shufflevector(alo, ahi,
                    0,1,2,3,4,5,6,7,8,9,10,11,12,13,14,15);

        const _Float16* wp = Wr + k0 + hi * 16;
        v8h b0lo = *(const v8h*)(wp);
        v8h b0hi = *(const v8h*)(wp + 8);
        v16h bv0 = __builtin_shufflevector(b0lo, b0hi,
                    0,1,2,3,4,5,6,7,8,9,10,11,12,13,14,15);
        acc0 = __builtin_amdgcn_wmma_f32_16x16x32_f16(
                   false, av, false, bv0, (short)0, acc0, false, false);

        wp += wjstride;
        v8h b1lo = *(const v8h*)(wp);
        v8h b1hi = *(const v8h*)(wp + 8);
        v16h bv1 = __builtin_shufflevector(b1lo, b1hi,
                    0,1,2,3,4,5,6,7,8,9,10,11,12,13,14,15);
        acc1 = __builtin_amdgcn_wmma_f32_16x16x32_f16(
                   false, av, false, bv1, (short)0, acc1, false, false);

        wp += wjstride;
        v8h b2lo = *(const v8h*)(wp);
        v8h b2hi = *(const v8h*)(wp + 8);
        v16h bv2 = __builtin_shufflevector(b2lo, b2hi,
                    0,1,2,3,4,5,6,7,8,9,10,11,12,13,14,15);
        acc2 = __builtin_amdgcn_wmma_f32_16x16x32_f16(
                   false, av, false, bv2, (short)0, acc2, false, false);

        wp += wjstride;
        v8h b3lo = *(const v8h*)(wp);
        v8h b3hi = *(const v8h*)(wp + 8);
        v16h bv3 = __builtin_shufflevector(b3lo, b3hi,
                    0,1,2,3,4,5,6,7,8,9,10,11,12,13,14,15);
        acc3 = __builtin_amdgcn_wmma_f32_16x16x32_f16(
                   false, av, false, bv3, (short)0, acc3, false, false);
    }

    float* Cm = C + blockIdx.z * sCm;
    v8f accs[4] = {acc0, acc1, acc2, acc3};
#pragma unroll
    for (int j = 0; j < 4; ++j) {
        int col = n0 + j * 16 + l;
        if (col < Nreal) {                      // store-only guard (post-WMMA)
            float bvv = bias ? bias[col] : 0.f;
#pragma unroll
            for (int r = 0; r < 8; ++r) {
                long row = m0 + r + hi * 8;
                long idx = row * (long)ldc + col;
                float v = accs[j][r] + bvv;
                if (add_res) v += Cm[idx];
                Cm[idx] = v;
            }
        }
    }
}

// ---------------- causal depthwise conv (k=4) + SiLU, f16 out ----------------
// grid = NMAM*B_*8 (8 chunks of 64 timesteps), block = 288 (264 active + pad writers)
__global__ void conv_silu(const float* __restrict__ xz,
                          const float* __restrict__ convw, long cw_mstride,
                          const float* __restrict__ convb, long cb_mstride,
                          _Float16* __restrict__ xih) {
    int blk = blockIdx.x;
    int c  = blk & 7;
    int mb = blk >> 3;
    int b  = mb % B_;
    int m  = mb / B_;
    int d  = threadIdx.x;
    long rowbase = (long)m * RM + (long)b * L_;
    int t0 = c * 64;

    if (d >= DI) {
        for (int t = t0; t < t0 + 64; ++t)
            xih[(rowbase + t) * KP_DI + d] = (_Float16)0.f;
        return;
    }
    const float* w = convw + (long)m * cw_mstride + (long)d * 4;
    float w0 = w[0], w1 = w[1], w2 = w[2], w3 = w[3];
    float bias = convb[(long)m * cb_mstride + d];

    float u0 = 0.f, u1 = 0.f, u2 = 0.f;
    if (t0 >= 3) u0 = xz[(rowbase + t0 - 3) * NR_IN + d];
    if (t0 >= 2) u1 = xz[(rowbase + t0 - 2) * NR_IN + d];
    if (t0 >= 1) u2 = xz[(rowbase + t0 - 1) * NR_IN + d];

    for (int t = t0; t < t0 + 64; ++t) {
        float u3 = xz[(rowbase + t) * NR_IN + d];
        float a = w0 * u0 + w1 * u1 + w2 * u2 + w3 * u3 + bias;
        float s = a / (1.f + __expf(-a));           // SiLU
        xih[(rowbase + t) * KP_DI + d] = (_Float16)s;
        u0 = u1; u1 = u2; u2 = u3;
    }
}

// ---------------- fused dt-proj + softplus + selective scan + gate ----------------
// grid = NMAM*B_, block = 288 (264 active channels)
__global__ void scan_kernel(const float* __restrict__ dbc,
                            const _Float16* __restrict__ xih,
                            const float* __restrict__ xz,
                            const float* __restrict__ dtw, long dtw_ms,
                            const float* __restrict__ dtb, long dtb_ms,
                            const float* __restrict__ Alog, long A_ms,
                            const float* __restrict__ Dp, long D_ms,
                            _Float16* __restrict__ ygh) {
    __shared__ float sh[48];                        // dt[9] | B[16] | C[16]
    int b = blockIdx.x % B_;
    int m = blockIdx.x / B_;
    int d = threadIdx.x;
    long rowbase = (long)m * RM + (long)b * L_;
    bool active = d < DI;

    float wdt[DTR], Ar[DS], h[DS];
    float bdt = 0.f, Dd = 0.f;
    if (active) {
#pragma unroll
        for (int r = 0; r < DTR; ++r) wdt[r] = dtw[(long)m * dtw_ms + (long)d * DTR + r];
        bdt = dtb[(long)m * dtb_ms + d];
#pragma unroll
        for (int n = 0; n < DS; ++n) Ar[n] = -__expf(Alog[(long)m * A_ms + (long)d * DS + n]);
        Dd = Dp[(long)m * D_ms + d];
#pragma unroll
        for (int n = 0; n < DS; ++n) h[n] = 0.f;
    }

    for (int t = 0; t < L_; ++t) {
        long row = rowbase + t;
        if (t + 1 < L_) {   // hide next-step latency on the sequential chain
            __builtin_prefetch(&dbc[(row + 1) * NR_XP], 0, 0);
            __builtin_prefetch(&xih[(row + 1) * KP_DI + d], 0, 0);
            __builtin_prefetch(&xz[(row + 1) * NR_IN + DI + d], 0, 0);
        }
        if (d < DTR + 2 * DS) sh[d] = dbc[row * NR_XP + d];
        __syncthreads();
        float out = 0.f;
        if (active) {
            float x = bdt;
#pragma unroll
            for (int r = 0; r < DTR; ++r) x += sh[r] * wdt[r];
            float delta = (x > 20.f) ? x : log1pf(__expf(x));   // softplus
            float xi = (float)xih[row * KP_DI + d];
            float acc = 0.f;
#pragma unroll
            for (int n = 0; n < DS; ++n) {
                float dA = __expf(delta * Ar[n]);
                h[n] = dA * h[n] + (delta * sh[DTR + n]) * xi;
                acc += h[n] * sh[DTR + DS + n];
            }
            float y = acc + Dd * xi;
            float z = xz[row * NR_IN + DI + d];
            float g = z / (1.f + __expf(-z));                   // SiLU(z)
            out = y * g;
        }
        ygh[row * KP_DI + d] = (_Float16)out;                   // pads get 0
        __syncthreads();
    }
}

// ---------------- per-(m,l) 8x8 cross-batch attention ----------------
// grid = NMAM*L_, block = 64 (48 active: head x batch)
__global__ void mha_kernel(const float* __restrict__ qkv, float* __restrict__ ctx) {
    int blk = blockIdx.x;
    int l = blk % L_;
    int m = blk / L_;
    int tid = threadIdx.x;
    if (tid >= NH * B_) return;
    int hh = tid / B_;
    int s  = tid % B_;
    const float* base = qkv + (long)m * RM * NR_QKV;
    const float scale = rsqrtf((float)HD);

    float q[HD];
#pragma unroll
    for (int d0 = 0; d0 < HD; ++d0)
        q[d0] = base[((long)s * L_ + l) * NR_QKV + hh * HD + d0] * scale;

    float sc[B_];
    float mx = -1e30f;
#pragma unroll
    for (int t = 0; t < B_; ++t) {
        const float* kr = base + ((long)t * L_ + l) * NR_QKV + DM + hh * HD;
        float a = 0.f;
#pragma unroll
        for (int d0 = 0; d0 < HD; ++d0) a += q[d0] * kr[d0];
        sc[t] = a;
        mx = fmaxf(mx, a);
    }
    float se = 0.f;
#pragma unroll
    for (int t = 0; t < B_; ++t) { sc[t] = __expf(sc[t] - mx); se += sc[t]; }
    float inv = 1.f / se;

    float* crow = ctx + ((long)m * RM + (long)s * L_ + l) * DM + hh * HD;
#pragma unroll
    for (int d0 = 0; d0 < HD; ++d0) {
        float a = 0.f;
#pragma unroll
        for (int t = 0; t < B_; ++t)
            a += sc[t] * base[((long)t * L_ + l) * NR_QKV + 2 * DM + hh * HD + d0];
        crow[d0] = a * inv;
    }
}

// ---------------- mean over L ----------------
// grid = NMAM*B_, block = 160
__global__ void mean_ctx(const float* __restrict__ ctx, float* __restrict__ feats) {
    int b = blockIdx.x % B_;
    int m = blockIdx.x / B_;
    int d = threadIdx.x;
    if (d >= DM) return;
    const float* base = ctx + ((long)m * RM + (long)b * L_) * DM;
    float s = 0.f;
    for (int l = 0; l < L_; ++l) s += base[(long)l * DM + d];
    feats[((long)m * B_ + b) * DM + d] = s * (1.f / L_);
}

// ---------------- head: attn out-proj on pooled feats + static MLP + classifier ----
// single block, 256 threads
__global__ void head_kernel(const float* __restrict__ feats,
                            const float* __restrict__ wo, const float* __restrict__ bo,
                            const float* __restrict__ stat,
                            const float* __restrict__ sw1, const float* __restrict__ sb1,
                            const float* __restrict__ sg1, const float* __restrict__ sbe1,
                            const float* __restrict__ sw2, const float* __restrict__ sb2,
                            const float* __restrict__ sg2, const float* __restrict__ sbe2,
                            const float* __restrict__ sw3, const float* __restrict__ sb3,
                            const float* __restrict__ rw,  const float* __restrict__ rb,
                            const float* __restrict__ rg,  const float* __restrict__ rbe,
                            const float* __restrict__ cw1, const float* __restrict__ cb1,
                            const float* __restrict__ cw2, const float* __restrict__ cb2,
                            const float* __restrict__ cw3, const float* __restrict__ cb3,
                            float* __restrict__ outp) {
    __shared__ float comb[B_][788];
    __shared__ float s1[B_][256];
    __shared__ float s2[B_][128];
    __shared__ float h1[B_][32];
    __shared__ float h2[B_][16];
    int tid = threadIdx.x;
    const float bnscale = rsqrtf(1.f + 1e-5f);

    // ts_ens: comb[b][m*132+e] = feats[m][b] . wo[e] + bo[e]
    for (int i = tid; i < B_ * NMAM * DM; i += 256) {
        int b = i / (NMAM * DM);
        int r = i % (NMAM * DM);
        int m = r / DM;
        int e = r % DM;
        const float* f = feats + ((long)m * B_ + b) * DM;
        const float* w = wo + (long)e * DM;
        float a = bo[e];
        for (int d0 = 0; d0 < DM; ++d0) a += f[d0] * w[d0];
        comb[b][m * DM + e] = a;
    }
    __syncthreads();

    // s1 = relu(bn(static @ se_w1^T + se_b1))
    for (int i = tid; i < B_ * 256; i += 256) {
        int b = i / 256, j = i % 256;
        const float* w = sw1 + (long)j * 64;
        float a = sb1[j];
        for (int k = 0; k < 64; ++k) a += stat[b * 64 + k] * w[k];
        a = a * (sg1[j] * bnscale) + sbe1[j];
        s1[b][j] = fmaxf(a, 0.f);
    }
    __syncthreads();

    // s2 = relu(bn(s1 @ se_w2^T + se_b2))
    for (int i = tid; i < B_ * 128; i += 256) {
        int b = i / 128, j = i % 128;
        const float* w = sw2 + (long)j * 256;
        float a = sb2[j];
        for (int k = 0; k < 256; ++k) a += s1[b][k] * w[k];
        a = a * (sg2[j] * bnscale) + sbe2[j];
        s2[b][j] = fmaxf(a, 0.f);
    }
    __syncthreads();

    // s = relu(bn((s2 @ se_w3^T + se_b3) + (static @ res_w^T + res_b)))
    for (int i = tid; i < B_ * 128; i += 256) {
        int b = i / 128, j = i % 128;
        const float* w3 = sw3 + (long)j * 128;
        float a = sb3[j];
        for (int k = 0; k < 128; ++k) a += s2[b][k] * w3[k];
        const float* wr = rw + (long)j * 64;
        float r = rb[j];
        for (int k = 0; k < 64; ++k) r += stat[b * 64 + k] * wr[k];
        float v = (a + r) * (rg[j] * bnscale) + rbe[j];
        comb[b][660 + j] = fmaxf(v, 0.f);
    }
    __syncthreads();

    // h1 = relu(comb @ c_w1^T + c_b1)
    if (tid < B_ * 32) {
        int b = tid / 32, j = tid % 32;
        const float* w = cw1 + (long)j * 788;
        float a = cb1[j];
        for (int k = 0; k < 788; ++k) a += comb[b][k] * w[k];
        h1[b][j] = fmaxf(a, 0.f);
    }
    __syncthreads();

    // h2 = relu(h1 @ c_w2^T + c_b2)
    if (tid < B_ * 16) {
        int b = tid / 16, j = tid % 16;
        const float* w = cw2 + (long)j * 32;
        float a = cb2[j];
        for (int k = 0; k < 32; ++k) a += h1[b][k] * w[k];
        h2[b][j] = fmaxf(a, 0.f);
    }
    __syncthreads();

    // out = h2 @ c_w3^T + c_b3
    if (tid < B_ * 2) {
        int b = tid / 2, o = tid % 2;
        const float* w = cw3 + (long)o * 16;
        float a = cb3[o];
        for (int k = 0; k < 16; ++k) a += h2[b][k] * w[k];
        outp[b * 2 + o] = a;
    }
}

// ============================= host launcher =============================
extern "C" void kernel_launch(void* const* d_in, const int* in_sizes, int n_in,
                              void* d_out, int out_size, void* d_ws, size_t ws_size,
                              hipStream_t stream) {
    (void)in_sizes; (void)n_in; (void)out_size; (void)ws_size;
    const float* x         = (const float*)d_in[0];
    const float* stat      = (const float*)d_in[1];
    /* d_in[2] = time, unused by the reference forward */
    const float* mask      = (const float*)d_in[3];
    const float* norm_w    = (const float*)d_in[4];   // (5,4,132)
    const float* in_proj   = (const float*)d_in[5];   // (5,4,528,132)
    const float* conv_w    = (const float*)d_in[6];   // (5,4,264,1,4)
    const float* conv_b    = (const float*)d_in[7];   // (5,4,264)
    const float* x_proj    = (const float*)d_in[8];   // (5,4,41,264)
    const float* dt_w      = (const float*)d_in[9];   // (5,4,264,9)
    const float* dt_b      = (const float*)d_in[10];  // (5,4,264)
    const float* A_log     = (const float*)d_in[11];  // (5,4,264,16)
    const float* Dp        = (const float*)d_in[12];  // (5,4,264)
    const float* out_proj  = (const float*)d_in[13];  // (5,4,132,264)
    const float* attn_in_w = (const float*)d_in[14];  // (396,132)
    const float* attn_in_b = (const float*)d_in[15];  // (396)
    const float* attn_ow   = (const float*)d_in[16];  // (132,132)
    const float* attn_ob   = (const float*)d_in[17];  // (132)
    const float* se_w1 = (const float*)d_in[18]; const float* se_b1 = (const float*)d_in[19];
    const float* se_g1 = (const float*)d_in[20]; const float* se_be1= (const float*)d_in[21];
    const float* se_w2 = (const float*)d_in[22]; const float* se_b2 = (const float*)d_in[23];
    const float* se_g2 = (const float*)d_in[24]; const float* se_be2= (const float*)d_in[25];
    const float* se_w3 = (const float*)d_in[26]; const float* se_b3 = (const float*)d_in[27];
    const float* res_w = (const float*)d_in[28]; const float* res_b = (const float*)d_in[29];
    const float* res_g = (const float*)d_in[30]; const float* res_be= (const float*)d_in[31];
    const float* c_w1  = (const float*)d_in[32]; const float* c_b1  = (const float*)d_in[33];
    const float* c_w2  = (const float*)d_in[34]; const float* c_b2  = (const float*)d_in[35];
    const float* c_w3  = (const float*)d_in[36]; const float* c_b3  = (const float*)d_in[37];

    // ---------------- workspace carve ----------------
    char* base = (char*)d_ws;
    size_t off = 0;
    auto carve = [&](size_t bytes) -> void* {
        void* p = base + off;
        off = (off + bytes + 255) & ~(size_t)255;
        return p;
    };
    float*    act  = (float*)   carve((size_t)ROWS * DM * 4);
    _Float16* xnh  = (_Float16*)carve((size_t)ROWS * KP_DM * 2);
    float*    xz   = (float*)   carve((size_t)ROWS * NR_IN * 4);
    _Float16* xih  = (_Float16*)carve((size_t)ROWS * KP_DI * 2);
    float*    dbc  = (float*)   carve((size_t)ROWS * NR_XP * 4);
    _Float16* ygh  = (_Float16*)carve((size_t)ROWS * KP_DI * 2);
    _Float16* wIn  = (_Float16*)carve((size_t)20 * NP_IN * KP_DM * 2);
    _Float16* wXp  = (_Float16*)carve((size_t)20 * NP_XP * KP_DI * 2);
    _Float16* wOp  = (_Float16*)carve((size_t)20 * NP_OP * KP_DI * 2);
    _Float16* wAi  = (_Float16*)carve((size_t)NP_QKV * KP_DM * 2);
    float*    bAi  = (float*)   carve((size_t)NP_QKV * 4);
    float*    ctx  = (float*)   carve((size_t)ROWS * DM * 4);
    float*    feats= (float*)   carve((size_t)NMAM * B_ * DM * 4);
    float*    qkv  = xz;   // xz region is free after layers; 5*4096*400*4 < 5*4096*528*4

    // ---------------- pack weights (per call; deterministic) ----------------
    pack_w_f16<<<7200, 256, 0, stream>>>(in_proj,  wIn, 20, NR_IN, NP_IN, DM, KP_DM);
    pack_w_f16<<<1440, 256, 0, stream>>>(x_proj,   wXp, 20, 41,    NP_XP, DI, KP_DI);
    pack_w_f16<<<4320, 256, 0, stream>>>(out_proj, wOp, 20, DM,    NP_OP, DI, KP_DI);
    pack_w_f16<<<280,  256, 0, stream>>>(attn_in_w,wAi, 1,  396,   NP_QKV,DM, KP_DM);
    pack_bias_f32<<<2, 256, 0, stream>>>(attn_in_b, bAi, 396, NR_QKV);
    init_act<<<10560, 256, 0, stream>>>(x, mask, act);

    // one wave per block; grid.y = exact number of 64-col strips (no early exits)
    auto gemm = [&](const _Float16* A, long sAm, const _Float16* W, long sWm,
                    float* C, long sCm, int ldc, int Npad64, int Nreal, int K,
                    const float* bias, int add_res) {
        dim3 grid(RM / 16, (unsigned)(Npad64 / 64), NMAM);
        gemm_wmma<<<grid, 32, 0, stream>>>(A, sAm, W, sWm, C, sCm,
                                           ldc, Nreal, K, bias, add_res);
    };

    // ---------------- 4 mamba layers (5 mambas in parallel via grid.z) ----------------
    for (int layer = 0; layer < NLAY; ++layer) {
        rmsnorm_pack<<<ROWS, KP_DM, 0, stream>>>(act, norm_w + layer * DM,
                                                 (long)NLAY * DM, xnh, 1);
        // xz = xn @ in_proj^T      (M=4096, Npad=576 real 528, K=160)
        gemm(xnh, (long)RM * KP_DM,
             wIn + (long)layer * NP_IN * KP_DM, (long)NLAY * NP_IN * KP_DM,
             xz, (long)RM * NR_IN, NR_IN, NP_IN, NR_IN, KP_DM, nullptr, 0);
        // causal depthwise conv + SiLU -> xih (f16, padded)
        conv_silu<<<NMAM * B_ * 8, KP_DI, 0, stream>>>(
            xz, conv_w + (long)layer * DI * 4, (long)NLAY * DI * 4,
            conv_b + (long)layer * DI, (long)NLAY * DI, xih);
        // dbc = xi @ x_proj^T      (M=4096, Npad=64 real 48, K=288)
        gemm(xih, (long)RM * KP_DI,
             wXp + (long)layer * NP_XP * KP_DI, (long)NLAY * NP_XP * KP_DI,
             dbc, (long)RM * NR_XP, NR_XP, NP_XP, NR_XP, KP_DI, nullptr, 0);
        // fused dt-proj + softplus + selective scan + D-skip + SiLU gate
        scan_kernel<<<NMAM * B_, KP_DI, 0, stream>>>(
            dbc, xih, xz,
            dt_w + (long)layer * DI * DTR, (long)NLAY * DI * DTR,
            dt_b + (long)layer * DI,       (long)NLAY * DI,
            A_log + (long)layer * DI * DS, (long)NLAY * DI * DS,
            Dp + (long)layer * DI,         (long)NLAY * DI, ygh);
        // act += yg @ out_proj^T   (M=4096, Npad=192 real 132, K=288)
        gemm(ygh, (long)RM * KP_DI,
             wOp + (long)layer * NP_OP * KP_DI, (long)NLAY * NP_OP * KP_DI,
             act, (long)RM * DM, DM, NP_OP, DM, KP_DI, nullptr, 1);
    }

    // ---------------- attention over batch, pooled head ----------------
    rmsnorm_pack<<<ROWS, KP_DM, 0, stream>>>(act, nullptr, 0, xnh, 0);
    gemm(xnh, (long)RM * KP_DM, wAi, 0,
         qkv, (long)RM * NR_QKV, NR_QKV, NP_QKV, NR_QKV, KP_DM, bAi, 0);
    mha_kernel<<<NMAM * L_, 64, 0, stream>>>(qkv, ctx);
    mean_ctx<<<NMAM * B_, 160, 0, stream>>>(ctx, feats);
    head_kernel<<<1, 256, 0, stream>>>(feats, attn_ow, attn_ob, stat,
                                       se_w1, se_b1, se_g1, se_be1,
                                       se_w2, se_b2, se_g2, se_be2,
                                       se_w3, se_b3, res_w, res_b, res_g, res_be,
                                       c_w1, c_b1, c_w2, c_b2, c_w3, c_b3,
                                       (float*)d_out);
}
